// RefinerTransformerEdgeClassifier_53455162966539
// MI455X (gfx1250) — matrix operand order repaired and checked by reading, not compiled, not run on hardware
//
#include <hip/hip_runtime.h>
#include <hip/hip_bf16.h>
#include <cstdint>
#include <cstddef>

// ---------------------------------------------------------------------------
// RefinerTransformerEdgeClassifier on gfx1250 (MI455X).
// GEMMs + attention on v_wmma_f32_16x16x32_f16; weight/K/V tiles staged into
// LDS with the CDNA5 async global->LDS path (ASYNCcnt) and double buffering.
// ---------------------------------------------------------------------------

#define DMODEL 512
#define FFDIM  2048
#define NLAYER 6
#define NHEAD  8
#define HDIM   64
#define BATCH  8
#define SEQ    1024
#define NROWS  (BATCH * SEQ)   // 8192

typedef _Float16 half16  __attribute__((ext_vector_type(16)));
typedef _Float16 half8   __attribute__((ext_vector_type(8)));
typedef float    floatx8 __attribute__((ext_vector_type(8)));

// ---- CDNA5 async global->LDS copy (guarded; sync LDS staging fallback) ----
#if defined(__has_builtin)
#  if __has_builtin(__builtin_amdgcn_global_load_async_to_lds_b128)
#    define HAVE_ASYNC_LDS 1
#  endif
#endif
#ifndef HAVE_ASYNC_LDS
#  define HAVE_ASYNC_LDS 0
#endif

// builtin signature (from hipcc diagnostic): pointee is vector_size(16) int
typedef int v4i_t __attribute__((vector_size(16)));
typedef __attribute__((address_space(1))) v4i_t* gv4_ptr;
typedef __attribute__((address_space(3))) v4i_t* lv4_ptr;

// copy 16 bytes (8 halfs) global -> LDS
__device__ __forceinline__ void stage16(const _Float16* g, _Float16* l) {
#if HAVE_ASYNC_LDS
  __builtin_amdgcn_global_load_async_to_lds_b128(
      (gv4_ptr)(uintptr_t)g, (lv4_ptr)(uint32_t)(uintptr_t)l, 0, 0);
#else
  *(half8*)l = *(const half8*)g;
#endif
}

__device__ __forceinline__ void stage_wait() {
#if HAVE_ASYNC_LDS
#  if defined(__has_builtin) && __has_builtin(__builtin_amdgcn_s_wait_asynccnt)
  __builtin_amdgcn_s_wait_asynccnt(0);
#  else
  asm volatile("s_wait_asynccnt 0x0" ::: "memory");
#  endif
#endif
}

// ---------------- WMMA fragment loaders (CDNA5 16x16x32 f16 layouts) -------
// A 16x32 (MxK) from row-major src: lane m=lane&15; lanes<16 K=k0+0..15,
// lanes>=16 K=k0+16..31 (two contiguous 8-half chunks).
__device__ __forceinline__ half16 load_a_frag(const _Float16* __restrict__ A,
                                              int lda, int row0, int k0, int lane) {
  int m  = lane & 15;
  int kb = (lane >> 4) << 4;
  const _Float16* p = A + (size_t)(row0 + m) * lda + k0 + kb;
  half8 lo = *(const half8*)(p);
  half8 hi = *(const half8*)(p + 8);
  half16 r;
#pragma unroll
  for (int i = 0; i < 8; ++i) { r[i] = lo[i]; r[i + 8] = hi[i]; }
  return r;
}

// B 32x16 (KxN) from row-major W (N x K): lane n=lane&15 holds col n,
// K = k0 + 16*(lane>=16) + 0..15 contiguous.
__device__ __forceinline__ half16 load_b_frag(const _Float16* __restrict__ W,
                                              int ldw, int n0, int k0, int lane) {
  int n   = lane & 15;
  int kb2 = (lane >> 4) << 4;
  return *(const half16*)(W + (size_t)(n0 + n) * ldw + k0 + kb2);
}

__device__ __forceinline__ floatx8 wmma_f16(half16 a, half16 b, floatx8 c) {
  return __builtin_amdgcn_wmma_f32_16x16x32_f16(false, a, false, b, (short)0,
                                                c, false, false);
}

// ---------------- elementwise f32 -> f16 -----------------------------------
__global__ void cvt_f32_f16_kernel(const float* __restrict__ in,
                                   _Float16* __restrict__ out, int n) {
  int i = blockIdx.x * blockDim.x + threadIdx.x;
  if (i < n) out[i] = (_Float16)in[i];
}

__global__ void maskbias_kernel(const unsigned char* __restrict__ kpm,
                                float* __restrict__ mb, int n) {
  int i = blockIdx.x * blockDim.x + threadIdx.x;
  if (i < n) mb[i] = kpm[i] ? -1.0e30f : 0.0f;
}

// ---------------- WMMA GEMM: C[M,N] = A[M,K] @ W[N,K]^T + bias -------------
// 128 threads = 4 waves; wave computes 32x64 (8 accumulators). B tile (64x32)
// staged into double-buffered LDS via async copy; A fragments reg-pipelined.
__global__ __launch_bounds__(128) void gemm_wmma_kernel(
    const _Float16* __restrict__ A, const _Float16* __restrict__ W,
    const float* __restrict__ bias, float* __restrict__ Cf,
    _Float16* __restrict__ Ch, int M, int N, int K, int relu) {
  __shared__ __align__(128) _Float16 bbuf[2][64 * 32];   // 2 x 4KB
  int lane = threadIdx.x & 31;
  int wave = threadIdx.x >> 5;
  int row0 = blockIdx.y * 128 + wave * 32;
  int col0 = blockIdx.x * 64;

  // B staging: thread t copies 32B: n = t>>1, k = (t&1)*16
  int ln = threadIdx.x >> 1;
  int lk = (threadIdx.x & 1) << 4;
  const _Float16* gW = W + (size_t)(col0 + ln) * K + lk;
  _Float16* l0 = &bbuf[0][ln * 32 + lk];
  _Float16* l1 = &bbuf[1][ln * 32 + lk];

  int nsteps = K >> 5;
  // stage step 0 into buf 0
  stage16(gW, l0);
  stage16(gW + 8, l0 + 8);

  floatx8 acc[2][4] = {{{}, {}, {}, {}}, {{}, {}, {}, {}}};
  // register-pipelined A fragments for step 0
  half16 a0 = load_a_frag(A, K, row0, 0, lane);
  half16 a1 = load_a_frag(A, K, row0 + 16, 0, lane);

  for (int s = 0; s < nsteps; ++s) {
    int k0 = s << 5;
    stage_wait();          // our async fills of bbuf[s&1] done
    __syncthreads();       // whole tile filled; prev buffer free to overwrite
    if (s + 1 < nsteps) {  // uniform branch: stage next tile (in flight during wmma)
      _Float16* ld = (s & 1) ? l0 : l1;
      stage16(gW + k0 + 32, ld);
      stage16(gW + k0 + 40, ld + 8);
    }
    // prefetch next A fragments into registers (clamped -> always valid)
    int kn = (s + 1 < nsteps) ? k0 + 32 : k0;
    half16 na0 = load_a_frag(A, K, row0, kn, lane);
    half16 na1 = load_a_frag(A, K, row0 + 16, kn, lane);

    const _Float16* bb = bbuf[s & 1];
#pragma unroll
    for (int t = 0; t < 4; ++t) {
      half16 vb = load_b_frag(bb, 32, 16 * t, 0, lane);   // ds reads
      acc[0][t] = wmma_f16(a0, vb, acc[0][t]);
      acc[1][t] = wmma_f16(a1, vb, acc[1][t]);
    }
    a0 = na0; a1 = na1;
  }

  int rbase = (lane >> 4) << 3;
#pragma unroll
  for (int rt = 0; rt < 2; ++rt)
#pragma unroll
    for (int t = 0; t < 4; ++t) {
      int n = col0 + 16 * t + (lane & 15);
      float bv = bias ? bias[n] : 0.0f;
#pragma unroll
      for (int r = 0; r < 8; ++r) {
        float v = acc[rt][t][r] + bv;
        if (relu) v = v > 0.0f ? v : 0.0f;
        size_t idx = (size_t)(row0 + rt * 16 + rbase + r) * N + n;
        if (Cf) Cf[idx] = v;
        if (Ch) Ch[idx] = (_Float16)v;
      }
    }
}

// ---------------- V transpose per (b,h): (N x HD) -> (HD x N), via LDS -----
__global__ __launch_bounds__(1024) void vtrans_kernel(
    const _Float16* __restrict__ V, _Float16* __restrict__ VT) {
  __shared__ _Float16 tile[64][65];
  int bh = blockIdx.y;
  int n0 = blockIdx.x * 64;
  int b = bh >> 3, h = bh & 7;
  const _Float16* Vb = V + (size_t)b * SEQ * DMODEL + h * HDIM;
  int tx = threadIdx.x, ty = threadIdx.y;
#pragma unroll
  for (int k = 0; k < 4; ++k) {
    int n = ty + k * 16;
    tile[n][tx] = Vb[(size_t)(n0 + n) * DMODEL + tx];
  }
  __syncthreads();
  _Float16* VTb = VT + (size_t)bh * HDIM * SEQ;
#pragma unroll
  for (int k = 0; k < 4; ++k) {
    int d = ty + k * 16;
    VTb[(size_t)d * SEQ + n0 + tx] = tile[tx][d];
  }
}

// ---------------- flash attention ------------------------------------------
// 128 threads = 4 waves, all on the same (b,h); wave w handles query tile
// qt = (blockIdx&15)*4 + w. K-tile (32x64) and V^T-tile (64x32) per 32-key
// block are shared by the 4 waves: staged into double-buffered LDS (async).
// S^T = K_tile @ Q^T (keys on rows -> per-query reduce = 8 regs + 1 shuffle),
// online softmax, attn^T = V^T @ P^T.
__global__ __launch_bounds__(128) void attention_kernel(
    const _Float16* __restrict__ Q, const _Float16* __restrict__ Km,
    const _Float16* __restrict__ VT, const float* __restrict__ maskb,
    _Float16* __restrict__ Out) {
  __shared__ __align__(128) _Float16 kbuf[2][32 * 64];   // 2 x 4KB
  __shared__ __align__(128) _Float16 vbuf[2][64 * 32];   // 2 x 4KB
  int lane = threadIdx.x & 31;
  int wave = threadIdx.x >> 5;
  int bh = blockIdx.x >> 4;                    // 0..63 = b*8+h (block-uniform)
  int b = bh >> 3, h = bh & 7;
  int qt = ((blockIdx.x & 15) << 2) + wave;
  int qc = lane & 15;
  int hh = lane >> 4;
  const float scale = 0.125f;                  // 1/sqrt(64)

  const _Float16* Qb  = Q  + (size_t)b * SEQ * DMODEL + h * HDIM;
  const _Float16* Kb  = Km + (size_t)b * SEQ * DMODEL + h * HDIM;
  const _Float16* VTb = VT + (size_t)bh * HDIM * SEQ;
  int q0 = qt * 16;

  // staging indices: K-tile thread copies key kr, ch kc..kc+15;
  //                  V^T-tile thread copies ch vd, keys vk..vk+15.
  int tr = threadIdx.x;
  int kr = tr >> 2, kc = (tr & 3) << 4;
  int vd = tr >> 1, vk = (tr & 1) << 4;

  // Q as resident B-fragments (channels = WMMA K dim)
  half16 bq0 = load_b_frag(Qb, DMODEL, q0, 0, lane);
  half16 bq1 = load_b_frag(Qb, DMODEL, q0, 32, lane);

  float mi = -1.0e30f, li = 0.0f;
  floatx8 acc[4] = {{}, {}, {}, {}};           // attn^T: 64 ch x 16 queries

  // stage block 0
  {
    const _Float16* gk = Kb + (size_t)kr * DMODEL + kc;
    stage16(gk, &kbuf[0][kr * 64 + kc]);
    stage16(gk + 8, &kbuf[0][kr * 64 + kc + 8]);
    const _Float16* gv = VTb + (size_t)vd * SEQ + vk;
    stage16(gv, &vbuf[0][vd * 32 + vk]);
    stage16(gv + 8, &vbuf[0][vd * 32 + vk + 8]);
  }

  for (int jb = 0; jb < SEQ / 32; ++jb) {
    int kbase = jb * 32;
    stage_wait();
    __syncthreads();
    if (jb + 1 < SEQ / 32) {                   // uniform: stage next block
      int buf = (jb + 1) & 1;
      const _Float16* gk = Kb + (size_t)(kbase + 32 + kr) * DMODEL + kc;
      stage16(gk, &kbuf[buf][kr * 64 + kc]);
      stage16(gk + 8, &kbuf[buf][kr * 64 + kc + 8]);
      const _Float16* gv = VTb + (size_t)vd * SEQ + kbase + 32 + vk;
      stage16(gv, &vbuf[buf][vd * 32 + vk]);
      stage16(gv + 8, &vbuf[buf][vd * 32 + vk + 8]);
    }
    const _Float16* kt = kbuf[jb & 1];
    const _Float16* vt = vbuf[jb & 1];

    // S^T tiles (keys kbase..+15 and +16..+31) from LDS K-tile (lda=64)
    floatx8 st0 = {}, st1 = {};
    st0 = wmma_f16(load_a_frag(kt, 64, 0, 0, lane), bq0, st0);
    st0 = wmma_f16(load_a_frag(kt, 64, 0, 32, lane), bq1, st0);
    st1 = wmma_f16(load_a_frag(kt, 64, 16, 0, lane), bq0, st1);
    st1 = wmma_f16(load_a_frag(kt, 64, 16, 32, lane), bq1, st1);

    // scale + key-padding bias; this lane's keys: kbase + r + 8*hh (+16)
    const float* mb0 = maskb + b * SEQ + kbase + 8 * hh;
    float s0[8], s1[8], mnew = -1.0e30f;
#pragma unroll
    for (int r = 0; r < 8; ++r) {
      s0[r] = st0[r] * scale + mb0[r];
      s1[r] = st1[r] * scale + mb0[r + 16];
      mnew = fmaxf(mnew, fmaxf(s0[r], s1[r]));
    }
    mnew = fmaxf(mnew, __shfl_xor(mnew, 16, 32));
    float mcur = fmaxf(mi, mnew);
    float corr = __expf(mi - mcur);

    float p0[8], p1[8], lsum = 0.0f;
#pragma unroll
    for (int r = 0; r < 8; ++r) {
      p0[r] = __expf(s0[r] - mcur);
      p1[r] = __expf(s1[r] - mcur);
      lsum += p0[r] + p1[r];
    }
    lsum += __shfl_xor(lsum, 16, 32);
    li = li * corr + lsum;
    mi = mcur;
#pragma unroll
    for (int t = 0; t < 4; ++t)
#pragma unroll
      for (int r = 0; r < 8; ++r) acc[t][r] *= corr;

    // build P^T B-fragment (32 keys x 16 queries) via cross-half shuffles
    float p0x[8], p1x[8];
#pragma unroll
    for (int r = 0; r < 8; ++r) {
      p0x[r] = __shfl_xor(p0[r], 16, 32);
      p1x[r] = __shfl_xor(p1[r], 16, 32);
    }
    half16 pb;
#pragma unroll
    for (int r = 0; r < 8; ++r) {
      float lo = hh ? p1x[r] : p0[r];
      float hi = hh ? p1[r]  : p0x[r];
      pb[r]     = (_Float16)lo;
      pb[r + 8] = (_Float16)hi;
    }

    // attn^T += V^T_tile @ P^T from LDS V^T-tile (lda=32)
#pragma unroll
    for (int cb = 0; cb < 4; ++cb) {
      half16 av = load_a_frag(vt, 32, cb * 16, 0, lane);
      acc[cb] = wmma_f16(av, pb, acc[cb]);
    }
  }

  float inv = 1.0f / li;
  _Float16* ob = Out + (size_t)(b * SEQ + q0 + qc) * DMODEL + h * HDIM;
#pragma unroll
  for (int cb = 0; cb < 4; ++cb)
#pragma unroll
    for (int r = 0; r < 8; ++r)
      ob[cb * 16 + r + 8 * hh] = (_Float16)(acc[cb][r] * inv);
}

// ---------------- residual add + LayerNorm (D=512), one wave per row -------
__global__ __launch_bounds__(128) void ln_kernel(
    const float* __restrict__ x, const float* __restrict__ resid,
    const float* __restrict__ g, const float* __restrict__ be,
    float* __restrict__ yf, _Float16* __restrict__ yh) {
  int lane = threadIdx.x & 31;
  int row = blockIdx.x * 4 + (threadIdx.x >> 5);
  const float* xr = x + (size_t)row * DMODEL;
  const float* rr = resid ? resid + (size_t)row * DMODEL : nullptr;
  float v[16], s = 0.0f;
#pragma unroll
  for (int i = 0; i < 16; ++i) {
    float t = xr[lane + 32 * i];
    if (rr) t += rr[lane + 32 * i];
    v[i] = t; s += t;
  }
#pragma unroll
  for (int o = 16; o >= 1; o >>= 1) s += __shfl_xor(s, o, 32);
  float mu = s * (1.0f / DMODEL);
  float vs = 0.0f;
#pragma unroll
  for (int i = 0; i < 16; ++i) { float d = v[i] - mu; vs += d * d; }
#pragma unroll
  for (int o = 16; o >= 1; o >>= 1) vs += __shfl_xor(vs, o, 32);
  float rs = rsqrtf(vs * (1.0f / DMODEL) + 1e-5f);
#pragma unroll
  for (int i = 0; i < 16; ++i) {
    int c = lane + 32 * i;
    float yv = (v[i] - mu) * rs * g[c] + be[c];
    if (yf) yf[(size_t)row * DMODEL + c] = yv;
    if (yh) yh[(size_t)row * DMODEL + c] = (_Float16)yv;
  }
}

// ---------------- tiny head: logits[8192,5] --------------------------------
__global__ void head_kernel(const float* __restrict__ X,
                            const float* __restrict__ Wh,
                            const float* __restrict__ bh,
                            float* __restrict__ out) {
  int idx = blockIdx.x * blockDim.x + threadIdx.x;
  if (idx >= NROWS * 5) return;
  int row = idx / 5, c = idx % 5;
  const float* xr = X + (size_t)row * DMODEL;
  const float* wr = Wh + c * DMODEL;
  float s = bh[c];
  for (int k = 0; k < DMODEL; ++k) s += xr[k] * wr[k];
  out[idx] = s;
}

// ---------------------------------------------------------------------------
extern "C" void kernel_launch(void* const* d_in, const int* in_sizes, int n_in,
                              void* d_out, int out_size, void* d_ws, size_t ws_size,
                              hipStream_t stream) {
  (void)in_sizes; (void)n_in; (void)out_size; (void)ws_size;
  const float* src     = (const float*)d_in[0];
  const unsigned char* kpm = (const unsigned char*)d_in[1];
  const float* embed_W = (const float*)d_in[2];
  const float* embed_b = (const float*)d_in[3];
  const float* Wq = (const float*)d_in[4];  const float* bq = (const float*)d_in[5];
  const float* Wk = (const float*)d_in[6];  const float* bk = (const float*)d_in[7];
  const float* Wv = (const float*)d_in[8];  const float* bv = (const float*)d_in[9];
  const float* Wo = (const float*)d_in[10]; const float* bo = (const float*)d_in[11];
  const float* W1 = (const float*)d_in[12]; const float* b1 = (const float*)d_in[13];
  const float* W2 = (const float*)d_in[14]; const float* b2 = (const float*)d_in[15];
  const float* ln1_g = (const float*)d_in[16]; const float* ln1_b = (const float*)d_in[17];
  const float* ln2_g = (const float*)d_in[18]; const float* ln2_b = (const float*)d_in[19];
  const float* fin_g = (const float*)d_in[20]; const float* fin_b = (const float*)d_in[21];
  const float* head_W = (const float*)d_in[22]; const float* head_b = (const float*)d_in[23];
  float* logits = (float*)d_out;

  char* cur = (char*)d_ws;
  auto alloc = [&](size_t bytes) {
    char* p = cur;
    cur += (bytes + 255) & ~(size_t)255;
    return p;
  };
  float*    x_f32   = (float*)   alloc((size_t)NROWS * DMODEL * 4);
  _Float16* x_f16   = (_Float16*)alloc((size_t)NROWS * DMODEL * 2);
  float*    tmp_f32 = (float*)   alloc((size_t)NROWS * DMODEL * 4);
  _Float16* q_f16   = (_Float16*)alloc((size_t)NROWS * DMODEL * 2);
  _Float16* k_f16   = (_Float16*)alloc((size_t)NROWS * DMODEL * 2);
  _Float16* v_f16   = (_Float16*)alloc((size_t)NROWS * DMODEL * 2);
  _Float16* vT_f16  = (_Float16*)alloc((size_t)NROWS * DMODEL * 2);
  _Float16* at_f16  = (_Float16*)alloc((size_t)NROWS * DMODEL * 2);
  _Float16* hid_f16 = (_Float16*)alloc((size_t)NROWS * FFDIM * 2);
  _Float16* src_h   = (_Float16*)alloc((size_t)NROWS * 64 * 2);
  _Float16* eW_h    = (_Float16*)alloc((size_t)DMODEL * 64 * 2);
  _Float16* Wq_h    = (_Float16*)alloc((size_t)NLAYER * DMODEL * DMODEL * 2);
  _Float16* Wk_h    = (_Float16*)alloc((size_t)NLAYER * DMODEL * DMODEL * 2);
  _Float16* Wv_h    = (_Float16*)alloc((size_t)NLAYER * DMODEL * DMODEL * 2);
  _Float16* Wo_h    = (_Float16*)alloc((size_t)NLAYER * DMODEL * DMODEL * 2);
  _Float16* W1_h    = (_Float16*)alloc((size_t)NLAYER * FFDIM * DMODEL * 2);
  _Float16* W2_h    = (_Float16*)alloc((size_t)NLAYER * DMODEL * FFDIM * 2);
  float*    maskb   = (float*)   alloc((size_t)BATCH * SEQ * 4);

  auto cvt = [&](const float* in, _Float16* out, int n) {
    cvt_f32_f16_kernel<<<(n + 255) / 256, 256, 0, stream>>>(in, out, n);
  };
  cvt(src, src_h, NROWS * 64);
  cvt(embed_W, eW_h, DMODEL * 64);
  cvt(Wq, Wq_h, NLAYER * DMODEL * DMODEL);
  cvt(Wk, Wk_h, NLAYER * DMODEL * DMODEL);
  cvt(Wv, Wv_h, NLAYER * DMODEL * DMODEL);
  cvt(Wo, Wo_h, NLAYER * DMODEL * DMODEL);
  cvt(W1, W1_h, NLAYER * FFDIM * DMODEL);
  cvt(W2, W2_h, NLAYER * DMODEL * FFDIM);
  maskbias_kernel<<<(BATCH * SEQ + 255) / 256, 256, 0, stream>>>(kpm, maskb, BATCH * SEQ);

  auto gemm = [&](const _Float16* A, const _Float16* W, const float* bias,
                  float* Cf, _Float16* Ch, int M, int N, int K, int relu) {
    dim3 grid(N / 64, M / 128);
    gemm_wmma_kernel<<<grid, 128, 0, stream>>>(A, W, bias, Cf, Ch, M, N, K, relu);
  };

  // embed: x = src @ embed_W^T + embed_b
  gemm(src_h, eW_h, embed_b, x_f32, x_f16, NROWS, DMODEL, 64, 0);

  for (int l = 0; l < NLAYER; ++l) {
    const _Float16* wq = Wq_h + (size_t)l * DMODEL * DMODEL;
    const _Float16* wk = Wk_h + (size_t)l * DMODEL * DMODEL;
    const _Float16* wv = Wv_h + (size_t)l * DMODEL * DMODEL;
    const _Float16* wo = Wo_h + (size_t)l * DMODEL * DMODEL;
    const _Float16* w1 = W1_h + (size_t)l * FFDIM * DMODEL;
    const _Float16* w2 = W2_h + (size_t)l * DMODEL * FFDIM;

    gemm(x_f16, wq, bq + l * DMODEL, nullptr, q_f16, NROWS, DMODEL, DMODEL, 0);
    gemm(x_f16, wk, bk + l * DMODEL, nullptr, k_f16, NROWS, DMODEL, DMODEL, 0);
    gemm(x_f16, wv, bv + l * DMODEL, nullptr, v_f16, NROWS, DMODEL, DMODEL, 0);

    vtrans_kernel<<<dim3(SEQ / 64, BATCH * NHEAD), dim3(64, 16), 0, stream>>>(v_f16, vT_f16);
    attention_kernel<<<BATCH * NHEAD * (SEQ / 16) / 4, 128, 0, stream>>>(
        q_f16, k_f16, vT_f16, maskb, at_f16);

    gemm(at_f16, wo, bo + l * DMODEL, tmp_f32, nullptr, NROWS, DMODEL, DMODEL, 0);
    ln_kernel<<<NROWS / 4, 128, 0, stream>>>(
        x_f32, tmp_f32, ln1_g + l * DMODEL, ln1_b + l * DMODEL, x_f32, x_f16);

    gemm(x_f16, w1, b1 + l * FFDIM, nullptr, hid_f16, NROWS, FFDIM, DMODEL, 1);
    gemm(hid_f16, w2, b2 + l * DMODEL, tmp_f32, nullptr, NROWS, DMODEL, FFDIM, 0);
    ln_kernel<<<NROWS / 4, 128, 0, stream>>>(
        x_f32, tmp_f32, ln2_g + l * DMODEL, ln2_b + l * DMODEL, x_f32, x_f16);
  }

  ln_kernel<<<NROWS / 4, 128, 0, stream>>>(x_f32, nullptr, fin_g, fin_b, tmp_f32, nullptr);
  head_kernel<<<(NROWS * 5 + 255) / 256, 256, 0, stream>>>(tmp_f32, head_W, head_b, logits);
}
